// ConstraintPredictorGNN_41205916238042
// MI455X (gfx1250) — compile-verified, matrix operand order checked by reading, not run on hardware
//
#include <hip/hip_runtime.h>
#include <hip/hip_bf16.h>

typedef __attribute__((ext_vector_type(16))) __bf16 v16bf;
typedef __attribute__((ext_vector_type(8)))  float  v8f;

#define N_NODES 20000
#define N_EDGES 150000
#define N_PAIRS 500000

// ---- WMMA helper: D = A(16x32 bf16) * B(32x16 bf16) + C(f32) --------------
__device__ __forceinline__ v8f wmma_bf16(v16bf a, v16bf b, v8f c) {
    return __builtin_amdgcn_wmma_f32_16x16x32_bf16(false, a, false, b,
                                                   (short)0, c, false, false);
}

// K-offset inside a 32-wide K tile for 16-bit A/B fragments (ISA 7.12.2):
// lane-half Lhi (0/1), element j (0..15)
__device__ __forceinline__ int koffs(int lhi, int j) {
    return (j < 8) ? (lhi * 8 + j) : (16 + lhi * 8 + (j - 8));
}
// inverse: kc (0..31) -> (lhi, j)
__device__ __forceinline__ void kinv(int kc, int& lhi, int& j) {
    if (kc < 16) { lhi = kc >> 3; j = kc & 7; }
    else         { int t = kc - 16; lhi = t >> 3; j = 8 + (t & 7); }
}

// ---- pack a KxN f32 matrix into per-lane-contiguous B fragments -----------
// dst[((kt*numNT + nt)*32 + L)*16 + j] = src[k*Nn + n]  (zero-padded OOB)
__global__ void pack_b_kernel(const float* __restrict__ src, __bf16* __restrict__ dst,
                              int K, int Nn, int numKT, int numNT) {
    int idx = blockIdx.x * blockDim.x + threadIdx.x;
    int total = numKT * numNT * 512;
    if (idx >= total) return;
    int j = idx & 15;
    int L = (idx >> 4) & 31;
    int rest = idx >> 9;
    int nt = rest % numNT;
    int kt = rest / numNT;
    int k = kt * 32 + koffs(L >> 4, j);
    int n = nt * 16 + (L & 15);
    float v = (k < K && n < Nn) ? src[k * Nn + n] : 0.0f;
    dst[idx] = (__bf16)v;
}

__global__ void f32_to_bf16_kernel(const float* __restrict__ src,
                                   __bf16* __restrict__ dst, int n) {
    int i = blockIdx.x * blockDim.x + threadIdx.x;
    if (i < n) dst[i] = (__bf16)src[i];
}

__global__ void zero_f32_kernel(float* __restrict__ p, int n) {
    int i = blockIdx.x * blockDim.x + threadIdx.x;
    if (i < n) p[i] = 0.0f;
}

__global__ void copy_u32_kernel(const unsigned int* __restrict__ src,
                                unsigned int* __restrict__ dst, int n) {
    int i = blockIdx.x * blockDim.x + threadIdx.x;
    if (i < n) dst[i] = src[i];
}

// ---------------------------------------------------------------------------
// Fused NNConv edge pipeline, 16 edges per block (256 threads = 8 waves):
//   H1  = relu(edge_attr @ w1 + b1)            (1 WMMA / n-tile, K padded to 32)
//   W_e = H1 @ w2 + b2                          (32 kt x 64 nt WMMAs, streamed)
//   msg = x[src] . W_e   folded per D-tile into LDS (ds_add_f32)
//   agg[dst] += msg                             (global_atomic_add_f32)
// Phase C uses kt-outer / nt-inner with 8 accumulators per wave so one LDS
// A-fragment feeds 8 consecutive WMMAs.
// ---------------------------------------------------------------------------
__global__ void __launch_bounds__(256)
nnconv_edge_kernel(const float* __restrict__ edge_attr,   // [E,16]
                   const int*   __restrict__ edge_index,  // [2,E]
                   const float* __restrict__ xin,         // [N,32] f32 input feats
                   const __bf16* __restrict__ w1p,        // packed [64][32][16]
                   const float* __restrict__ b1,          // [1024]
                   const __bf16* __restrict__ w2p,        // packed [32][64][32][16]
                   const float* __restrict__ b2,          // [1024]
                   float* __restrict__ agg,               // [N,32] (atomic accum)
                   int E)
{
    __shared__ __align__(32) __bf16 sA1p[32][16];       // A frag, K padded to 32
    __shared__ __align__(32) __bf16 sH1p[32][32][16];   // H1 in A-frag order [kt][L][j]
    __shared__ __align__(16) float sX[16][32];          // x[src] tile (f32)
    __shared__ __align__(16) float sMsg[16][32];        // message accumulator

    const int tid  = threadIdx.x;
    const int lane = tid & 31;
    const int wave = tid >> 5;
    const int e0   = blockIdx.x * 16;

    // ---- Phase A: stage tiles ----
    for (int s = tid; s < 512; s += 256) {              // sA1p (edge_attr, K pad)
        int L = s >> 4, j = s & 15;
        float v = 0.0f;
        if (j < 8) {                                    // koffs<16 iff j<8 (kt=0)
            int f = (L >> 4) * 8 + j;                   // 0..15
            v = edge_attr[(e0 + (L & 15)) * 16 + f];
        }
        sA1p[L][j] = (__bf16)v;
    }
    if (tid < 128) {                                    // gather x[src], float4
        int m = tid >> 3, c4 = tid & 7;
        int node = edge_index[e0 + m];                  // src row
        ((float4*)&sX[m][0])[c4] = ((const float4*)(xin + node * 32))[c4];
        ((float4*)&sMsg[0][0])[tid] = make_float4(0.f, 0.f, 0.f, 0.f);
    }
    __syncthreads();

    // ---- Phase B: H1 = relu(A @ w1 + b1) -> sH1p (A-fragment order) ----
    v16bf afrag = *(const v16bf*)&sA1p[lane][0];
    const int n  = lane & 15;
    const int mb = (lane >> 4) * 8;
    for (int q = 0; q < 8; ++q) {
        int nt = wave * 8 + q;
        v16bf bfrag = *(const v16bf*)&w1p[(nt * 32 + lane) * 16];
        v8f acc = {};
        acc = wmma_bf16(afrag, bfrag, acc);
        int col = nt * 16 + n;
        float bias = b1[col];
        int kt = col >> 5, lhi, j;
        kinv(col & 31, lhi, j);
        #pragma unroll
        for (int r = 0; r < 8; ++r) {
            float v = acc[r] + bias;
            v = v > 0.0f ? v : 0.0f;
            sH1p[kt][lhi * 16 + (mb + r)][j] = (__bf16)v;
        }
    }
    __syncthreads();

    // ---- Phase C: W_e tiles = H1 @ w2 + b2, folded into msg ----
    // kt outer: one A fragment (LDS) feeds 8 WMMAs; B streams from L2.
    v8f acc[8] = {};
    for (int kt = 0; kt < 32; ++kt) {
        v16bf a = *(const v16bf*)&sH1p[kt][lane][0];
        const __bf16* brow = &w2p[((kt * 64 + wave * 8) * 32 + lane) * 16];
        #pragma unroll
        for (int q = 0; q < 8; ++q) {
            v16bf b = *(const v16bf*)&brow[q * 512];    // 32 lanes * 16 elems
            acc[q] = wmma_bf16(a, b, acc[q]);
        }
    }
    #pragma unroll
    for (int q = 0; q < 8; ++q) {
        int nt = wave * 8 + q;
        int i  = nt >> 1;                 // input-channel index of this col block
        int ob = (nt & 1) * 16;           // output-channel base
        float bias = b2[nt * 16 + n];
        #pragma unroll
        for (int r = 0; r < 8; ++r) {
            int m = mb + r;               // edge within tile
            float w = acc[q][r] + bias;   // W_e[i][ob+n] for edge m
            atomicAdd(&sMsg[m][ob + n], sX[m][i] * w);
        }
    }
    __syncthreads();

    // ---- Phase D: scatter msg to agg[dst] ----
    for (int s = tid; s < 512; s += 256) {
        int m = s >> 5, c = s & 31;
        int node = edge_index[E + e0 + m];              // dst row
        atomicAdd(&agg[node * 32 + c], sMsg[m][c]);
    }
}

// ---------------------------------------------------------------------------
// Node update: h = relu(x @ root + agg + bias); one 16-row tile per wave.
// ---------------------------------------------------------------------------
__global__ void __launch_bounds__(256)
node_update_kernel(const __bf16* __restrict__ xb,     // [N,32] bf16
                   const __bf16* __restrict__ rootp,  // packed [2][32][16]
                   const float* __restrict__ agg,     // [N,32]
                   const float* __restrict__ bias,    // [32]
                   float* __restrict__ hout,          // [N,32] f32
                   __bf16* __restrict__ hbout,        // [N,32] bf16
                   int Nn)
{
    int wave = threadIdx.x >> 5, lane = threadIdx.x & 31;
    int mt = blockIdx.x * 8 + wave;
    if (mt * 16 >= Nn) return;
    int row0 = mt * 16;
    int m = lane & 15, lhi = lane >> 4;

    union { v16bf v; unsigned int u[8]; } A;
    const unsigned int* xrow = (const unsigned int*)(xb + (row0 + m) * 32);
    #pragma unroll
    for (int p = 0; p < 4; ++p) {
        A.u[p]     = xrow[lhi * 4 + p];        // K = lhi*8 + 2p  (+1)
        A.u[p + 4] = xrow[8 + lhi * 4 + p];    // K = 16 + lhi*8 + 2p (+1)
    }
    #pragma unroll
    for (int nt = 0; nt < 2; ++nt) {
        v16bf b = *(const v16bf*)&rootp[(nt * 32 + lane) * 16];
        v8f acc = {};
        acc = wmma_bf16(A.v, b, acc);
        int col = nt * 16 + (lane & 15);
        int mbase = (lane >> 4) * 8;
        float bc = bias[col];
        #pragma unroll
        for (int r = 0; r < 8; ++r) {
            int row = row0 + mbase + r;
            float v = acc[r] + agg[row * 32 + col] + bc;
            v = v > 0.0f ? v : 0.0f;
            hout[row * 32 + col]  = v;
            hbout[row * 32 + col] = (__bf16)v;
        }
    }
}

// ---------------------------------------------------------------------------
// Candidate MLP: logits = relu(concat(h[a],h[b]) @ w1 + b1) @ w2 + b2
// 16 candidates per wave; layer2 uses N-padded (16) B with cols>=4 zero.
// ---------------------------------------------------------------------------
__global__ void __launch_bounds__(256)
cand_mlp_kernel(const int* __restrict__ cand,       // [P,2]
                const __bf16* __restrict__ hb,      // [N,32] bf16
                const __bf16* __restrict__ w1p,     // packed [2][2][32][16]
                const float* __restrict__ b1,       // [32]
                const __bf16* __restrict__ w2p,     // packed [32][16] (cols 4..15 = 0)
                const float* __restrict__ b2,       // [4]
                float* __restrict__ out,            // [P,4]
                int P)
{
    __shared__ __align__(32) __bf16 sZ[8][32][16];
    int wave = threadIdx.x >> 5, lane = threadIdx.x & 31;
    int pt = blockIdx.x * 8 + wave;
    int p0 = pt * 16;
    bool active = (p0 < P);
    int m = lane & 15, lhi = lane >> 4;
    int mbase = lhi * 8;

    if (active) {
        int ia = cand[(p0 + m) * 2 + 0];
        int ib = cand[(p0 + m) * 2 + 1];
        union { v16bf v; unsigned int u[8]; } A0, A1;
        const unsigned int* ra = (const unsigned int*)(hb + ia * 32);
        const unsigned int* rb = (const unsigned int*)(hb + ib * 32);
        #pragma unroll
        for (int p = 0; p < 4; ++p) {
            A0.u[p] = ra[lhi * 4 + p];  A0.u[p + 4] = ra[8 + lhi * 4 + p];
            A1.u[p] = rb[lhi * 4 + p];  A1.u[p + 4] = rb[8 + lhi * 4 + p];
        }
        #pragma unroll
        for (int nt = 0; nt < 2; ++nt) {
            v8f acc = {};
            acc = wmma_bf16(A0.v, *(const v16bf*)&w1p[((0 * 2 + nt) * 32 + lane) * 16], acc);
            acc = wmma_bf16(A1.v, *(const v16bf*)&w1p[((1 * 2 + nt) * 32 + lane) * 16], acc);
            int col = nt * 16 + m;
            float bc = b1[col];
            int zlhi, zj; kinv(col, zlhi, zj);
            #pragma unroll
            for (int r = 0; r < 8; ++r) {
                float v = acc[r] + bc;
                v = v > 0.0f ? v : 0.0f;
                sZ[wave][zlhi * 16 + (mbase + r)][zj] = (__bf16)v;
            }
        }
    }
    __syncthreads();
    if (active) {
        v16bf zfrag = *(const v16bf*)&sZ[wave][lane][0];
        v8f acc = {};
        acc = wmma_bf16(zfrag, *(const v16bf*)&w2p[lane * 16], acc);
        if (m < 4) {
            float bc = b2[m];
            #pragma unroll
            for (int r = 0; r < 8; ++r)
                out[(p0 + mbase + r) * 4 + m] = acc[r] + bc;
        }
    }
}

// ---------------------------------------------------------------------------
extern "C" void kernel_launch(void* const* d_in, const int* in_sizes, int n_in,
                              void* d_out, int out_size, void* d_ws, size_t ws_size,
                              hipStream_t stream) {
    const float* x         = (const float*)d_in[0];
    const int*   edge_idx  = (const int*)  d_in[1];
    const float* edge_attr = (const float*)d_in[2];
    const int*   cand      = (const int*)  d_in[3];
    const float* c1_w1 = (const float*)d_in[4];  const float* c1_b1 = (const float*)d_in[5];
    const float* c1_w2 = (const float*)d_in[6];  const float* c1_b2 = (const float*)d_in[7];
    const float* c1_rt = (const float*)d_in[8];  const float* c1_bs = (const float*)d_in[9];
    const float* c2_w1 = (const float*)d_in[10]; const float* c2_b1 = (const float*)d_in[11];
    const float* c2_w2 = (const float*)d_in[12]; const float* c2_b2 = (const float*)d_in[13];
    const float* c2_rt = (const float*)d_in[14]; const float* c2_bs = (const float*)d_in[15];
    const float* m_w1  = (const float*)d_in[16]; const float* m_b1  = (const float*)d_in[17];
    const float* m_w2  = (const float*)d_in[18]; const float* m_b2  = (const float*)d_in[19];

    float* out_logits = (float*)d_out;                       // [P,4]
    unsigned int* out_cand = (unsigned int*)d_out + 2000000; // [P,2] bit-copy
    float* out_h = (float*)d_out + 3000000;                  // [N,32]

    // ---- workspace carve-up (256B aligned) ----
    char* p = (char*)d_ws;
    auto take = [&](size_t bytes) { char* q = p; p += (bytes + 255) & ~(size_t)255; return q; };
    __bf16* c1w1p = (__bf16*)take(64 * 512 * 2);        // [64][32][16]
    __bf16* c2w1p = (__bf16*)take(64 * 512 * 2);
    __bf16* c1w2p = (__bf16*)take(32 * 64 * 512 * 2);   // [32][64][32][16]
    __bf16* c2w2p = (__bf16*)take(32 * 64 * 512 * 2);
    __bf16* c1rtp = (__bf16*)take(2 * 512 * 2);
    __bf16* c2rtp = (__bf16*)take(2 * 512 * 2);
    __bf16* mw1p  = (__bf16*)take(4 * 512 * 2);
    __bf16* mw2p  = (__bf16*)take(512 * 2);
    __bf16* xb    = (__bf16*)take((size_t)N_NODES * 32 * 2);
    float*  h1    = (float*) take((size_t)N_NODES * 32 * 4);
    __bf16* h1b   = (__bf16*)take((size_t)N_NODES * 32 * 2);
    __bf16* h2b   = (__bf16*)take((size_t)N_NODES * 32 * 2);
    float*  agg1  = (float*) take((size_t)N_NODES * 32 * 4);
    float*  agg2  = (float*) take((size_t)N_NODES * 32 * 4);
    (void)ws_size; (void)in_sizes; (void)n_in; (void)out_size;

    // ---- pack weights into WMMA B-fragment layout (bf16) ----
    pack_b_kernel<<<128, 256, 0, stream>>>(c1_w1, c1w1p, 16, 1024, 1, 64);
    pack_b_kernel<<<128, 256, 0, stream>>>(c2_w1, c2w1p, 16, 1024, 1, 64);
    pack_b_kernel<<<4096, 256, 0, stream>>>(c1_w2, c1w2p, 1024, 1024, 32, 64);
    pack_b_kernel<<<4096, 256, 0, stream>>>(c2_w2, c2w2p, 1024, 1024, 32, 64);
    pack_b_kernel<<<4, 256, 0, stream>>>(c1_rt, c1rtp, 32, 32, 1, 2);
    pack_b_kernel<<<4, 256, 0, stream>>>(c2_rt, c2rtp, 32, 32, 1, 2);
    pack_b_kernel<<<8, 256, 0, stream>>>(m_w1, mw1p, 64, 32, 2, 2);
    pack_b_kernel<<<2, 256, 0, stream>>>(m_w2, mw2p, 32, 4, 1, 1);
    f32_to_bf16_kernel<<<2500, 256, 0, stream>>>(x, xb, N_NODES * 32);
    zero_f32_kernel<<<2500, 256, 0, stream>>>(agg1, N_NODES * 32);
    zero_f32_kernel<<<2500, 256, 0, stream>>>(agg2, N_NODES * 32);

    // ---- layer 1 ----
    nnconv_edge_kernel<<<N_EDGES / 16, 256, 0, stream>>>(
        edge_attr, edge_idx, x, c1w1p, c1_b1, c1w2p, c1_b2, agg1, N_EDGES);
    node_update_kernel<<<157, 256, 0, stream>>>(
        xb, c1rtp, agg1, c1_bs, h1, h1b, N_NODES);

    // ---- layer 2 ----
    nnconv_edge_kernel<<<N_EDGES / 16, 256, 0, stream>>>(
        edge_attr, edge_idx, h1, c2w1p, c2_b1, c2w2p, c2_b2, agg2, N_EDGES);
    node_update_kernel<<<157, 256, 0, stream>>>(
        h1b, c2rtp, agg2, c2_bs, out_h, h2b, N_NODES);

    // ---- candidate MLP + candidates passthrough ----
    cand_mlp_kernel<<<3907, 256, 0, stream>>>(
        cand, h2b, mw1p, m_b1, mw2p, m_b2, out_logits, N_PAIRS);
    copy_u32_kernel<<<3907, 256, 0, stream>>>(
        (const unsigned int*)cand, out_cand, N_PAIRS * 2);
}
